// AnaphoricityScorer_61607010894571
// MI455X (gfx1250) — compile-verified, matrix operand order checked by reading, not run on hardware
//
#include <hip/hip_runtime.h>
#include <hip/hip_bf16.h>

// ---------------------------------------------------------------------------
// AnaphoricityScorer for MI455X (gfx1250, wave32, WMMA bf16 path)
//   rows = 512*50 = 25600, K = 3136 = 98*32, hid = 1024
//   GEMM1: 164 GFLOP -> v_wmma_f32_16x16x32_bf16, f32 accumulate
//   GEMM2 (hid->1) fused into epilogue.
//   512-thread WG (16 waves, 4/SIMD). Wave = 2 M-tiles x 4 N-tiles (64 acc
//   VGPRs). Cross-K-step double-buffering implemented as ping-pong with the
//   K loop unrolled by 2, so buffer rotation is register renaming (no movs).
// ---------------------------------------------------------------------------

typedef __attribute__((ext_vector_type(16))) __bf16 v16bf;
typedef __attribute__((ext_vector_type(8)))  __bf16 v8bf;
typedef __attribute__((ext_vector_type(8)))  float  v8f;

#define EMB      1024
#define PW       64
#define INF      3136      // 3*EMB + PW
#define HID      1024
#define NANTS    50
#define BATCH    512
#define ROWS     (BATCH * NANTS)     // 25600
#define KSTEPS   98                  // 3136 / 32
#define MROWS    32                  // rows per workgroup
#define PITCH    3144                // halves per LDS row (3136 + 8 pad)
#define NTW      4                   // N-tiles per wave (16 waves x 4 = 64)
#define EPSILON  1e-07f

__device__ __forceinline__ unsigned short f2bf(float x) {
  union { float f; unsigned u; } v; v.f = x;
  unsigned r = v.u + 0x7FFFu + ((v.u >> 16) & 1u);   // round-to-nearest-even
  return (unsigned short)(r >> 16);
}

// ---------------------------------------------------------------------------
// Pre-pass: W1 (f32, [K=3136][N=1024] row-major) -> bf16 B-operand tiles.
// Tile (ntg, ks) = 512 bf16, lane l's v16bf contiguous at l*16:
// element e holds W1[k][n], n = ntg*16 + (l&15),
// k = ks*32 + (e&7) + ((l&16)?8:0) + ((e&8)?16:0)   (WMMA 16-bit B layout).
// ---------------------------------------------------------------------------
__global__ void prep_w1_kernel(const float* __restrict__ W1,
                               unsigned short* __restrict__ Wt) {
  int t    = blockIdx.x * 256 + threadIdx.x;     // 0 .. 3,211,263
  int e    = t & 15;
  int l    = (t >> 4) & 31;
  int tile = t >> 9;
  int ks   = tile % KSTEPS;
  int ntg  = tile / KSTEPS;
  int n    = ntg * 16 + (l & 15);
  int k    = ks * 32 + (e & 7) + ((l & 16) ? 8 : 0) + ((e & 8) ? 16 : 0);
  Wt[t] = f2bf(W1[(size_t)k * HID + n]);
}

// Dummy epsilon column: out[b][0] = 1e-7
__global__ void eps_kernel(float* __restrict__ out) {
  int b = blockIdx.x * 256 + threadIdx.x;
  if (b < BATCH) out[(size_t)b * (NANTS + 1)] = EPSILON;
}

// ---------------------------------------------------------------------------
// Main fused kernel: one WG = 32 rows, 512 threads (16 waves).
// ---------------------------------------------------------------------------
__global__ void __launch_bounds__(512, 1)
scorer_kernel(const float* __restrict__ allm,      // [2000][1024]
              const float* __restrict__ ments,     // [512][1024]
              const float* __restrict__ pw,        // [512][50][64]
              const int*   __restrict__ topIdx,    // [512][50]
              const float* __restrict__ rough,     // [512][50]
              const unsigned short* __restrict__ Wt, // bf16 tiles
              const float* __restrict__ b1,        // [1024]
              const float* __restrict__ W2,        // [1024]
              const float* __restrict__ b2,        // [1]
              float* __restrict__ out) {           // [512][51]
  extern __shared__ unsigned short smem_u[];
  unsigned short* tileA   = smem_u;                            // MROWS * PITCH halves
  float*          fineAcc = (float*)(smem_u + MROWS * PITCH);  // MROWS floats

  const int tid = threadIdx.x;

  // ---- Build bf16 pair-feature tile in LDS: [a | b | a*b | pw] ----
  {
    int row = tid >> 4;                             // 0..31 (16 threads/row)
    int c   = tid & 15;
    int rg  = blockIdx.x * MROWS + row;             // global row
    int b   = rg / NANTS;
    int midx = topIdx[rg];
    const float* amp = ments + (size_t)b * EMB;
    const float* bmp = allm + (size_t)midx * EMB;
    unsigned short* dst = tileA + row * PITCH;
#pragma unroll 4
    for (int j = 0; j < 64; ++j) {
      int f = c + j * 16;
      float av = amp[f];
      float bv = bmp[f];
      dst[f]            = f2bf(av);
      dst[EMB + f]      = f2bf(bv);
      dst[2 * EMB + f]  = f2bf(av * bv);
    }
    const float* pwp = pw + (size_t)rg * PW;
#pragma unroll
    for (int j = 0; j < 4; ++j) {
      int f = c + j * 16;
      dst[3 * EMB + f] = f2bf(pwp[f]);
    }
  }
  if (tid < MROWS) fineAcc[tid] = 0.0f;
  __syncthreads();

  // ---- WMMA GEMM: h = pair @ W1 ----
  const int lane = tid & 31;
  const int wave = tid >> 5;           // 0..15
  const int lrow = lane & 15;
  const int hsel = (lane >> 4) & 1;    // 0: K 0..7/16..23, 1: K 8..15/24..31

  // A-fragment loader: two 16B LDS runs per lane (K base, K base+16)
  auto loadA = [&](int mt, int ks) -> v16bf {
    const unsigned short* p = tileA + (mt * 16 + lrow) * PITCH + ks * 32 + hsel * 8;
    v8bf lo  = *(const v8bf*)(p);
    v8bf hi8 = *(const v8bf*)(p + 16);
    return __builtin_shufflevector(lo, hi8, 0, 1, 2, 3, 4, 5, 6, 7,
                                   8, 9, 10, 11, 12, 13, 14, 15);
  };

  // Per-lane base into this wave's 4 N-tile columns of Wt
  const unsigned short* wbase =
      Wt + (((size_t)wave * NTW * KSTEPS) << 9) + ((size_t)lane << 4);
  auto loadB = [&](int nt, int ks) -> v16bf {
    return *(const v16bf*)(wbase + (((size_t)nt * KSTEPS + ks) << 9));
  };

  v8f acc[2][NTW];
#pragma unroll
  for (int mt = 0; mt < 2; ++mt)
#pragma unroll
    for (int nt = 0; nt < NTW; ++nt)
      acc[mt][nt] = (v8f){0.f, 0.f, 0.f, 0.f, 0.f, 0.f, 0.f, 0.f};

  auto mmaStep = [&](const v16bf* bfr, v16bf aa0, v16bf aa1) {
#pragma unroll
    for (int nt = 0; nt < NTW; ++nt) {
      acc[0][nt] = __builtin_amdgcn_wmma_f32_16x16x32_bf16(
          false, aa0, false, bfr[nt], (short)0, acc[0][nt], false, false);
      acc[1][nt] = __builtin_amdgcn_wmma_f32_16x16x32_bf16(
          false, aa1, false, bfr[nt], (short)0, acc[1][nt], false, false);
    }
  };

  // Ping-pong buffers: A-set computes while B-set loads, then swap.
  v16bf bA[NTW], bB[NTW];
  v16bf a0A, a1A, a0B, a1B;
#pragma unroll
  for (int nt = 0; nt < NTW; ++nt) bA[nt] = loadB(nt, 0);
  a0A = loadA(0, 0);
  a1A = loadA(1, 0);

  // Steady state: 48 iterations cover ks = 0..95; loads reach ks+2 <= 96.
  for (int ks = 0; ks < KSTEPS - 2; ks += 2) {
    // phase 1: load ks+1 into B-set, compute ks from A-set
#pragma unroll
    for (int nt = 0; nt < NTW; ++nt) bB[nt] = loadB(nt, ks + 1);
    a0B = loadA(0, ks + 1);
    a1B = loadA(1, ks + 1);
    mmaStep(bA, a0A, a1A);
    // phase 2: load ks+2 into A-set, compute ks+1 from B-set
#pragma unroll
    for (int nt = 0; nt < NTW; ++nt) bA[nt] = loadB(nt, ks + 2);
    a0A = loadA(0, ks + 2);
    a1A = loadA(1, ks + 2);
    mmaStep(bB, a0B, a1B);
  }
  // Peeled tail: A-set holds ks=96; load ks=97 then compute both.
#pragma unroll
  for (int nt = 0; nt < NTW; ++nt) bB[nt] = loadB(nt, KSTEPS - 1);
  a0B = loadA(0, KSTEPS - 1);
  a1B = loadA(1, KSTEPS - 1);
  mmaStep(bA, a0A, a1A);
  mmaStep(bB, a0B, a1B);

  // ---- Fused epilogue: leaky_relu(h + b1) @ W2, reduced into LDS ----
  float partial[2][8];
#pragma unroll
  for (int mt = 0; mt < 2; ++mt)
#pragma unroll
    for (int j = 0; j < 8; ++j) partial[mt][j] = 0.0f;

#pragma unroll
  for (int nt = 0; nt < NTW; ++nt) {
    const int n = (wave * NTW + nt) * 16 + lrow;   // hidden column of this lane
    const float bias = b1[n];
    const float w2v  = W2[n];
#pragma unroll
    for (int mt = 0; mt < 2; ++mt) {
#pragma unroll
      for (int j = 0; j < 8; ++j) {
        float v = acc[mt][nt][j] + bias;
        v = (v > 0.0f) ? v : 0.01f * v;          // leaky_relu
        partial[mt][j] += v * w2v;
      }
    }
  }
  // D layout: lane holds N = lane&15; VGPR j -> row M = j + 8*(lane>=16)
  const int rbase = (lane & 16) >> 1;            // 0 or 8
#pragma unroll
  for (int mt = 0; mt < 2; ++mt)
#pragma unroll
    for (int j = 0; j < 8; ++j)
      atomicAdd(&fineAcc[mt * 16 + rbase + j], partial[mt][j]);

  __syncthreads();

  if (tid < MROWS) {
    int rg = blockIdx.x * MROWS + tid;
    int b  = rg / NANTS;
    int a  = rg - b * NANTS;
    out[(size_t)b * (NANTS + 1) + 1 + a] = rough[rg] + fineAcc[tid] + b2[0];
  }
}

// ---------------------------------------------------------------------------
extern "C" void kernel_launch(void* const* d_in, const int* in_sizes, int n_in,
                              void* d_out, int out_size, void* d_ws, size_t ws_size,
                              hipStream_t stream) {
  const float* allm  = (const float*)d_in[0];
  const float* ments = (const float*)d_in[1];
  const float* pw    = (const float*)d_in[2];
  const int*   tidx  = (const int*)d_in[3];
  const float* rough = (const float*)d_in[4];
  const float* W1    = (const float*)d_in[5];
  const float* b1    = (const float*)d_in[6];
  const float* W2    = (const float*)d_in[7];
  const float* b2    = (const float*)d_in[8];
  float* out = (float*)d_out;

  unsigned short* Wt = (unsigned short*)d_ws;   // 3136*1024*2 = 6.1 MB

  // 1) swizzle W1 -> bf16 WMMA B-operand tiles
  prep_w1_kernel<<<(INF * HID) / 256, 256, 0, stream>>>(W1, Wt);
  // 2) epsilon dummy column
  eps_kernel<<<(BATCH + 255) / 256, 256, 0, stream>>>(out);
  // 3) fused scorer
  const size_t smem = (size_t)MROWS * PITCH * sizeof(unsigned short)
                    + (size_t)MROWS * sizeof(float);   // ~201.3 KB (< 320 KB/WG)
  static_assert(MROWS * PITCH * 2 + MROWS * 4 < 320 * 1024, "LDS budget");
  hipFuncSetAttribute((const void*)scorer_kernel,
                      hipFuncAttributeMaxDynamicSharedMemorySize, (int)smem);
  scorer_kernel<<<ROWS / MROWS, 512, smem, stream>>>(
      allm, ments, pw, tidx, rough, Wt, b1, W2, b2, out);
}
